// HGNER_22625887716042
// MI455X (gfx1250) — matrix operand order repaired and checked by reading, not compiled, not run on hardware
//
#include <hip/hip_runtime.h>
#include <hip/hip_bf16.h>
#include <math.h>

#ifndef __has_builtin
#define __has_builtin(x) 0
#endif

// ---------------------------------------------------------------------------
// Problem constants (from reference): B=8, S=256, D=768, H=384, labels=9,
// windows {3,5,7}.  BS = B*S = 2048 independent window-LSTM sequences.
// ---------------------------------------------------------------------------
#define BS    2048
#define DIM   768
#define HID   384
#define G4    1536            // 4*HID gate width
#define NLBL  9
#define KSTEP 32              // K depth per WMMA

typedef __attribute__((ext_vector_type(16))) _Float16 v16h;
typedef __attribute__((ext_vector_type(8)))  float    v8f;

union FragU { v16h v; uint4 u[2]; };

// A-matrix (16x32 f16, MxK) fragment per CDNA5 ISA 7.12.2:
//   lane<16 : M=lane,    halves0-7 = K(k0+0..7),  halves8-15 = K(k0+16..23)
//   lane>=16: M=lane-16, halves0-7 = K(k0+8..15), halves8-15 = K(k0+24..31)
__device__ __forceinline__ v16h load_a_frag(const _Float16* p) {
    FragU r;
    r.u[0] = *(const uint4*)(p);
    r.u[1] = *(const uint4*)(p + 16);
    return r.v;
}
// B-matrix (32x16 f16, KxN) fragment (dense analog of the ISA sparse B table):
//   lane<16 : N=lane,    halves0-7 = K(k0+0..7),  halves8-15 = K(k0+16..23)
//   lane>=16: N=lane-16, halves0-7 = K(k0+8..15), halves8-15 = K(k0+24..31)
// Two 16B chunks per lane (p, p+16) from an N-major, K-contiguous row.
__device__ __forceinline__ v16h load_b_frag(const _Float16* p) {
    FragU r;
    r.u[0] = *(const uint4*)(p);
    r.u[1] = *(const uint4*)(p + 16);
    return r.v;
}

// ---------------------------------------------------------------------------
// Async global->LDS staging (CDNA5 GLOBAL_LOAD_ASYNC_TO_LDS_B128, ASYNCcnt).
// Inline asm per bridge doc (portable across toolchains).  LDS operand is the
// wave-relative byte offset = low 32 bits of the generic shared-memory addr.
// ---------------------------------------------------------------------------
#define USE_ASYNC_LDS 1

__device__ __forceinline__ void lds_copy16(const _Float16* g, _Float16* l) {
#if USE_ASYNC_LDS
    unsigned int       lo = (unsigned int)(unsigned long long)l;   // LDS byte addr
    unsigned long long ga = (unsigned long long)g;                 // 64-bit global
    asm volatile("global_load_async_to_lds_b128 %0, %1, off"
                 :: "v"(lo), "v"(ga) : "memory");
#else
    *(uint4*)l = *(const uint4*)g;
#endif
}

__device__ __forceinline__ void wait_async_lds() {
#if USE_ASYNC_LDS
#if __has_builtin(__builtin_amdgcn_s_wait_asynccnt)
    __builtin_amdgcn_s_wait_asynccnt(0);
#else
    asm volatile("s_wait_asynccnt 0x0" ::: "memory");
#endif
#endif
}

// ---------------------------------------------------------------------------
// C[M,N] = A[M,K] * B[N,K]^T  (A,B f16 row-major K-contiguous, C f32).
// Block = 4 waves = 128x64 output tile.  Wave w owns M rows [w*32,+32) as two
// 16-row tiles and all 64 N columns -> 8 accumulators; each K-step reuses 2 A
// fragments x 4 B fragments for 8 WMMAs.  The 64x32 B panel per K-step is
// staged to LDS with async b128 loads; the K-loop is hand-unrolled x2 with
// explicit ping/pong buffers so accumulators keep stable registers (no
// pipelining moves) and the next stage overlaps the current 8 WMMAs.
// blockIdx.z selects the (A,B,C) instance via element offsets.
// ---------------------------------------------------------------------------
template<int K>
__global__ void __launch_bounds__(128)
gemm_bt_wmma(const _Float16* __restrict__ A, long aOffZ,
             const _Float16* __restrict__ B, long bOff0, long bStepZ,
             float* __restrict__ C, long cOffZ, int N)
{
    A += (long)blockIdx.z * aOffZ;
    B += bOff0 + (long)blockIdx.z * bStepZ;
    C += (long)blockIdx.z * cOffZ;

    __shared__ __align__(16) _Float16 sB0[64 * KSTEP];   // ping (4KB)
    __shared__ __align__(16) _Float16 sB1[64 * KSTEP];   // pong (4KB)

    const int lane = threadIdx.x;            // 0..31
    const int wave = threadIdx.y;            // 0..3
    const int ftid = wave * 32 + lane;       // 0..127
    const int tn0  = blockIdx.y * 64;
    const int tm   = blockIdx.x * 128 + wave * 32;
    const int mrow0 = tm + (lane & 15);
    const _Float16* pa0 = A + (long)mrow0 * K + ((lane >> 4) << 3);
    const _Float16* pa1 = pa0 + 16 * K;

    // B panel staging: 64 rows x 64B = 256 x 16B chunks; 2 chunks per thread.
    const int ch0 = ftid * 2, ch1 = ch0 + 1;
    const int rn0 = ch0 >> 2, ko0 = (ch0 & 3) * 8;       // row, half-offset
    const int rn1 = ch1 >> 2, ko1 = (ch1 & 3) * 8;
    const _Float16* gb0 = B + (long)(tn0 + rn0) * K + ko0;
    const _Float16* gb1 = B + (long)(tn0 + rn1) * K + ko1;
    _Float16* l00 = &sB0[rn0 * KSTEP + ko0];
    _Float16* l01 = &sB0[rn1 * KSTEP + ko1];
    _Float16* l10 = &sB1[rn0 * KSTEP + ko0];
    _Float16* l11 = &sB1[rn1 * KSTEP + ko1];

    const int fragOff = (lane & 15) * KSTEP + ((lane >> 4) << 3);
    const _Float16* lbr0 = sB0 + fragOff;
    const _Float16* lbr1 = sB1 + fragOff;

    v8f acc0 = {}, acc1 = {}, acc2 = {}, acc3 = {};
    v8f acc4 = {}, acc5 = {}, acc6 = {}, acc7 = {};

#define GEMM_COMPUTE(lbr, ks)                                                              \
    do {                                                                                   \
        v16h a0 = load_a_frag(pa0 + (ks) * KSTEP);                                         \
        v16h a1 = load_a_frag(pa1 + (ks) * KSTEP);                                         \
        v16h b0 = load_b_frag((lbr));                                                      \
        v16h b1 = load_b_frag((lbr) + 16 * KSTEP);                                         \
        v16h b2 = load_b_frag((lbr) + 32 * KSTEP);                                         \
        v16h b3 = load_b_frag((lbr) + 48 * KSTEP);                                         \
        acc0 = __builtin_amdgcn_wmma_f32_16x16x32_f16(false, a0, false, b0, (short)0, acc0, false, false); \
        acc1 = __builtin_amdgcn_wmma_f32_16x16x32_f16(false, a0, false, b1, (short)0, acc1, false, false); \
        acc2 = __builtin_amdgcn_wmma_f32_16x16x32_f16(false, a0, false, b2, (short)0, acc2, false, false); \
        acc3 = __builtin_amdgcn_wmma_f32_16x16x32_f16(false, a0, false, b3, (short)0, acc3, false, false); \
        acc4 = __builtin_amdgcn_wmma_f32_16x16x32_f16(false, a1, false, b0, (short)0, acc4, false, false); \
        acc5 = __builtin_amdgcn_wmma_f32_16x16x32_f16(false, a1, false, b1, (short)0, acc5, false, false); \
        acc6 = __builtin_amdgcn_wmma_f32_16x16x32_f16(false, a1, false, b2, (short)0, acc6, false, false); \
        acc7 = __builtin_amdgcn_wmma_f32_16x16x32_f16(false, a1, false, b3, (short)0, acc7, false, false); \
    } while (0)

    const int nsteps = K / KSTEP;            // 24 or 12 (even)

    // preamble: stage K-step 0 into ping
    lds_copy16(gb0, l00);
    lds_copy16(gb1, l01);

#pragma unroll 1
    for (int ks = 0; ks < nsteps; ks += 2) {
        // ---- even step: consume ping, prefetch ks+1 into pong ----
        wait_async_lds();
        __syncthreads();
        lds_copy16(gb0 + (ks + 1) * KSTEP, l10);
        lds_copy16(gb1 + (ks + 1) * KSTEP, l11);
        GEMM_COMPUTE(lbr0, ks);
        // ---- odd step: consume pong, prefetch ks+2 into ping ----
        wait_async_lds();
        __syncthreads();
        if (ks + 2 < nsteps) {
            lds_copy16(gb0 + (ks + 2) * KSTEP, l00);
            lds_copy16(gb1 + (ks + 2) * KSTEP, l01);
        }
        GEMM_COMPUTE(lbr1, ks + 1);
    }
#undef GEMM_COMPUTE

    // C/D f32 layout: lane<16 -> M=tm+v, lane>=16 -> M=tm+8+v ; N = base + lane&15
    const int r0  = tm + ((lane >> 4) << 3);
    const int col = tn0 + (lane & 15);
#pragma unroll
    for (int v = 0; v < 8; v++) {
        float* cr = C + (long)(r0 + v) * N + col;
        cr[0]  = acc0[v];
        cr[16] = acc1[v];
        cr[32] = acc2[v];
        cr[48] = acc3[v];
        float* cr2 = cr + 16 * N;
        cr2[0]  = acc4[v];
        cr2[16] = acc5[v];
        cr2[32] = acc6[v];
        cr2[48] = acc7[v];
    }
}

// ---------------------------------------------------------------------------
// Small helper kernels
// ---------------------------------------------------------------------------
__global__ void zero4_kernel(float4* __restrict__ p, long n) {
    long i = (long)blockIdx.x * blockDim.x + threadIdx.x;
    long stride = (long)gridDim.x * blockDim.x;
    const float4 z = make_float4(0.f, 0.f, 0.f, 0.f);
    for (; i < n; i += stride) p[i] = z;
}

__global__ void cvt_f16_kernel(const float* __restrict__ src, _Float16* __restrict__ dst, int n) {
    int i = blockIdx.x * blockDim.x + threadIdx.x;
    int stride = gridDim.x * blockDim.x;
    for (; i < n; i += stride) dst[i] = (_Float16)src[i];
}

__global__ void bias_sum_kernel(const float* __restrict__ b0, const float* __restrict__ b1,
                                float* __restrict__ dst, int n) {
    int i = blockIdx.x * blockDim.x + threadIdx.x;
    if (i < n) dst[i] = b0[i] + b1[i];
}

// Inclusive-scan of valid flags per batch row -> compacted index (or -1).
__global__ void __launch_bounds__(256)
scan_kernel(const int* __restrict__ valid, int* __restrict__ jj) {
    const int b = blockIdx.x, tid = threadIdx.x;
    __shared__ int sh[256];
    const int v = (valid[b * 256 + tid] == 1) ? 1 : 0;
    sh[tid] = v;
    __syncthreads();
    for (int off = 1; off < 256; off <<= 1) {
        int t = (tid >= off) ? sh[tid - off] : 0;
        __syncthreads();
        sh[tid] += t;
        __syncthreads();
    }
    jj[b * 256 + tid] = v ? (sh[tid] - 1) : -1;
}

// Scatter valid token rows into compacted x (f32) + xh (f16). x/xh pre-zeroed.
__global__ void __launch_bounds__(256)
scatter_kernel(const float* __restrict__ seq, const int* __restrict__ jj,
               float* __restrict__ x, _Float16* __restrict__ xh) {
    const int n = blockIdx.x;          // b*S + s
    const int j = jj[n];
    if (j < 0) return;
    const int b = n >> 8;
    const float* src = seq + (long)n * DIM;
    float*    dst  = x  + ((long)(b * 256 + j)) * DIM;
    _Float16* dsth = xh + ((long)(b * 256 + j)) * DIM;
    for (int d = threadIdx.x; d < DIM; d += 256) {
        float v = src[d];
        dst[d] = v;
        dsth[d] = (_Float16)v;
    }
}

__device__ __forceinline__ float sigf(float x) { return 1.0f / (1.0f + expf(-x)); }

// Fused LSTM gate/state update for one time step, both directions (blockIdx.z).
//   fwd (z=0): window element t  -> source position p = s + t - hw
//   bwd (z=1): reversed window   -> source position p = s + hw - t
// mask = p in [0,S): state updates only when masked (matches reference where()).
__global__ void __launch_bounds__(256)
lstm_gate_kernel(const float* __restrict__ G, int sFwd, int sBwd,
                 const float* __restrict__ R, const float* __restrict__ biasS,
                 float* __restrict__ h, float* __restrict__ c, _Float16* __restrict__ hh,
                 int t, int hw, int useR)
{
    const int dir = blockIdx.z;
    const int gid = blockIdx.x * 256 + threadIdx.x;    // 0 .. BS*HID-1
    const int n = gid / HID;
    const int j = gid - n * HID;
    const int s = n & 255;
    const int p = s + ((dir == 0) ? (t - hw) : (hw - t));
    if (p < 0 || p >= 256) return;                     // masked: keep h,c
    const int b   = n >> 8;
    const int set = (dir == 0) ? sFwd : sBwd;
    const float* Gr = G + ((long)set * BS + (b << 8) + p) * G4 + j;
    const float* bb = biasS + set * G4 + j;
    float gi = Gr[0]        + bb[0];
    float gf = Gr[HID]      + bb[HID];
    float gg = Gr[2 * HID]  + bb[2 * HID];
    float go = Gr[3 * HID]  + bb[3 * HID];
    if (useR) {
        const float* Rr = R + ((long)dir * BS + n) * G4 + j;
        gi += Rr[0]; gf += Rr[HID]; gg += Rr[2 * HID]; go += Rr[3 * HID];
    }
    const long si = (long)dir * (BS * HID) + gid;
    float cv = c[si];
    float c2 = sigf(gf) * cv + sigf(gi) * tanhf(gg);
    float h2 = sigf(go) * tanhf(c2);
    c[si] = c2;
    h[si] = h2;
    hh[si] = (_Float16)h2;
}

// ml[b,s,win, dir*HID + j] = last hidden state (fwd then bwd concat).
__global__ void __launch_bounds__(256)
write_ml_kernel(const float* __restrict__ h, float* __restrict__ ml, int win) {
    const int dir = blockIdx.z;
    const int gid = blockIdx.x * 256 + threadIdx.x;
    const int n = gid / HID;
    const int j = gid - n * HID;
    ml[((long)n * 3 + win) * DIM + dir * HID + j] = h[(long)dir * (BS * HID) + gid];
}

// Per-position 3-way attention over window features + residual + 9-label head.
__global__ void __launch_bounds__(256)
attn_out_kernel(const float* __restrict__ x, const float* __restrict__ ml,
                const float* __restrict__ W, const float* __restrict__ bias,
                float* __restrict__ out)
{
    const int n = blockIdx.x;       // b*S + s
    const int tid = threadIdx.x;
    __shared__ float red[256];
    __shared__ float dsh[3];
    __shared__ float ash[3];
    const float* xr  = x  + (long)n * DIM;
    const float* mlr = ml + (long)n * 3 * DIM;

    float pv[3] = {0.f, 0.f, 0.f};
    for (int d = tid; d < DIM; d += 256) {
        float q = xr[d];
        pv[0] += q * mlr[d];
        pv[1] += q * mlr[DIM + d];
        pv[2] += q * mlr[2 * DIM + d];
    }
    for (int w = 0; w < 3; w++) {
        red[tid] = pv[w];
        __syncthreads();
        for (int st = 128; st > 0; st >>= 1) {
            if (tid < st) red[tid] += red[tid + st];
            __syncthreads();
        }
        if (tid == 0) dsh[w] = red[0];
        __syncthreads();
    }
    if (tid == 0) {
        const float inv = 0.03608439182435161f;  // 1/sqrt(768)
        float s0 = dsh[0] * inv, s1 = dsh[1] * inv, s2 = dsh[2] * inv;
        float m = fmaxf(s0, fmaxf(s1, s2));
        float e0 = expf(s0 - m), e1 = expf(s1 - m), e2 = expf(s2 - m);
        float is = 1.0f / (e0 + e1 + e2);
        ash[0] = e0 * is; ash[1] = e1 * is; ash[2] = e2 * is;
    }
    __syncthreads();
    const float a0 = ash[0], a1 = ash[1], a2 = ash[2];

    float lp[NLBL];
#pragma unroll
    for (int l = 0; l < NLBL; l++) lp[l] = 0.f;
    for (int d = tid; d < DIM; d += 256) {
        float y = xr[d] + a0 * mlr[d] + a1 * mlr[DIM + d] + a2 * mlr[2 * DIM + d];
#pragma unroll
        for (int l = 0; l < NLBL; l++) lp[l] += y * W[l * DIM + d];
    }
    for (int l = 0; l < NLBL; l++) {
        red[tid] = lp[l];
        __syncthreads();
        for (int st = 128; st > 0; st >>= 1) {
            if (tid < st) red[tid] += red[tid + st];
            __syncthreads();
        }
        if (tid == 0) out[(long)n * NLBL + l] = red[0] + bias[l];
        __syncthreads();
    }
}

// ---------------------------------------------------------------------------
// Host-side orchestration
// ---------------------------------------------------------------------------
extern "C" void kernel_launch(void* const* d_in, const int* in_sizes, int n_in,
                              void* d_out, int out_size, void* d_ws, size_t ws_size,
                              hipStream_t stream)
{
    (void)out_size; (void)ws_size;

    // ---- classify inputs by flat element count (robust to pytree order) ----
    const float* seq = nullptr;
    const float* Wlin = nullptr;
    const float* blin = nullptr;
    const int*   valid = nullptr;
    const float* Wih[6] = {};
    const float* Whh[6] = {};
    const float* bArr[12] = {};
    int nWih = 0, nWhh = 0, nB = 0;
    for (int i = 0; i < n_in; i++) {
        switch (in_sizes[i]) {
            case BS * DIM:        seq   = (const float*)d_in[i]; break;   // 1572864
            case NLBL * DIM:      Wlin  = (const float*)d_in[i]; break;   // 6912
            case NLBL:            blin  = (const float*)d_in[i]; break;   // 9
            case BS:              valid = (const int*)d_in[i];   break;   // 2048
            case G4 * DIM:        if (nWih < 6)  Wih[nWih++]  = (const float*)d_in[i]; break;
            case G4 * HID:        if (nWhh < 6)  Whh[nWhh++]  = (const float*)d_in[i]; break;
            case G4:              if (nB < 12)   bArr[nB++]   = (const float*)d_in[i]; break;
            default: break;
        }
    }
    // jax tree_flatten sorts dict keys: per window the 'bwd' param set comes
    // before 'fwd'.  Set appearance order: w0.bwd, w0.fwd, w1.bwd, w1.fwd, ...

    // ---- workspace layout (all region sizes are multiples of 256B) ----
    char* base = (char*)d_ws;
    size_t off = 0;
    auto take = [&](size_t bytes) -> char* { char* p = base + off; off += bytes; return p; };
    float*    x      = (float*)   take((size_t)BS * DIM * 4);            //  6.29 MB
    _Float16* xh     = (_Float16*)take((size_t)BS * DIM * 2);            //  3.15 MB (contiguous w/ x)
    _Float16* WihH   = (_Float16*)take((size_t)6 * G4 * DIM * 2);        // 14.16 MB
    _Float16* WhhH   = (_Float16*)take((size_t)6 * G4 * HID * 2);        //  7.08 MB
    float*    biasS  = (float*)   take((size_t)6 * G4 * 4);              //  0.04 MB
    float*    G      = (float*)   take((size_t)6 * BS * G4 * 4);         // 75.50 MB
    float*    R      = (float*)   take((size_t)2 * BS * G4 * 4);         // 25.17 MB
    float*    h      = (float*)   take((size_t)2 * BS * HID * 4);        //  6.29 MB
    float*    c      = (float*)   take((size_t)2 * BS * HID * 4);        //  6.29 MB (contig w/ h)
    _Float16* hh     = (_Float16*)take((size_t)2 * BS * HID * 2);        //  3.15 MB (contig w/ c)
    float*    ml     = (float*)   take((size_t)BS * 3 * DIM * 4);        // 18.87 MB
    int*      jj     = (int*)     take((size_t)BS * 4);

    // ---- 1) zero x|xh, scan valid_ids, scatter compacted rows ----
    zero4_kernel<<<2048, 256, 0, stream>>>((float4*)x,
        (long)((size_t)BS * DIM * 4 + (size_t)BS * DIM * 2) / 16);
    scan_kernel<<<8, 256, 0, stream>>>(valid, jj);
    scatter_kernel<<<BS, 256, 0, stream>>>(seq, jj, x, xh);

    // ---- 2) cast weights to f16, bias sums ----
    for (int s = 0; s < 6; s++) {
        cvt_f16_kernel<<<1024, 256, 0, stream>>>(Wih[s], WihH + (long)s * G4 * DIM, G4 * DIM);
        cvt_f16_kernel<<<1024, 256, 0, stream>>>(Whh[s], WhhH + (long)s * G4 * HID, G4 * HID);
        bias_sum_kernel<<<6, 256, 0, stream>>>(bArr[2 * s], bArr[2 * s + 1], biasS + s * G4, G4);
    }

    // ---- 3) input projections for all 6 LSTMs: G[set] = xh @ Wih[set]^T ----
    // M=2048, N=1536, K=768; grid: 16 (128-row) x 24 (64-col) x 6 sets
    gemm_bt_wmma<DIM><<<dim3(16, 24, 6), dim3(32, 4), 0, stream>>>(
        xh, 0L, WihH, 0L, (long)G4 * DIM, G, (long)BS * G4, G4);

    // ---- 4) window BiLSTMs ----
    const int Ls[3] = {3, 5, 7};
    for (int wi = 0; wi < 3; wi++) {
        const int L = Ls[wi], hw = L / 2;
        const int sBwd = 2 * wi, sFwd = 2 * wi + 1;   // sorted-key order: bwd first
        // zero h|c|hh (contiguous region)
        zero4_kernel<<<2048, 256, 0, stream>>>((float4*)h,
            (long)((size_t)2 * BS * HID * (4 + 4 + 2)) / 16);
        for (int t = 0; t < L; t++) {
            if (t > 0) {
                // R[dir] = hh[dir] @ Whh[set(dir)]^T   (M=2048,N=1536,K=384)
                gemm_bt_wmma<HID><<<dim3(16, 24, 2), dim3(32, 4), 0, stream>>>(
                    hh, (long)BS * HID,
                    WhhH, (long)sFwd * G4 * HID, (long)(sBwd - sFwd) * G4 * HID,
                    R, (long)BS * G4, G4);
            }
            lstm_gate_kernel<<<dim3((BS * HID) / 256, 1, 2), 256, 0, stream>>>(
                G, sFwd, sBwd, R, biasS, h, c, hh, t, hw, t > 0 ? 1 : 0);
        }
        write_ml_kernel<<<dim3((BS * HID) / 256, 1, 2), 256, 0, stream>>>(h, ml, wi);
    }

    // ---- 5) attention + residual + linear head ----
    attn_out_kernel<<<BS, 256, 0, stream>>>(x, ml, Wlin, blin, (float*)d_out);
}
